// DecoderRNN_47193100648594
// MI455X (gfx1250) — matrix operand order repaired and checked by reading, not compiled, not run on hardware
//
#include <hip/hip_runtime.h>

#define BATCH 10
#define HID 128
#define OUT 7
#define MAXLEN 2048
#define KDIM 160          // 32 (x slot, padded) + 128 (h)
#define GATES 512         // 4*HID

typedef __attribute__((ext_vector_type(16))) __bf16 v16bf;
typedef __attribute__((ext_vector_type(8)))  float  v8f;

union AFrag { unsigned int w[8]; v16bf v; };
union BFrag { uint4 q[2];        v16bf v; };

__device__ __forceinline__ float sigf(float x)      { return 1.0f / (1.0f + __expf(-x)); }
__device__ __forceinline__ float tanhfast(float x)  { return 2.0f / (1.0f + __expf(-2.0f * x)) - 1.0f; }

// LDS layout (elements)
#define WFRAG_N  (5*32*32*16)   // 81920 bf16 : Wcat B-fragments [kt][nt][lane][16]
#define WOFRAG_N (4*32*16)      //  2048 bf16 : W_out B-fragments [kt][lane][16]
#define Z_N      (16*KDIM)      //  2560 bf16 : A matrix rows [m][k]
#define SMEM_BYTES (((WFRAG_N + WOFRAG_N + Z_N) * 2) + (16*HID*2 + GATES + 256 + 16) * 4)

__global__ __launch_bounds__(256)
void decoder_lstm_wmma(const float* __restrict__ h0, const float* __restrict__ c0,
                       const float* __restrict__ tgt1h, const int* __restrict__ tf_mask,
                       const float* __restrict__ W_ih, const float* __restrict__ W_hh,
                       const float* __restrict__ b_ih, const float* __restrict__ b_hh,
                       const float* __restrict__ W_out, const float* __restrict__ b_out,
                       float* __restrict__ out)
{
    extern __shared__ char smem[];
    __bf16* Wfrag  = (__bf16*)smem;
    __bf16* WoFrag = Wfrag + WFRAG_N;
    __bf16* z      = WoFrag + WOFRAG_N;
    float*  cS     = (float*)(z + Z_N);
    float*  hS     = cS + 16*HID;
    float*  biasS  = hS + 16*HID;
    float*  logS   = biasS + GATES;     // 16x16 logits
    float*  boS    = logS + 256;        // 16 (b_out padded)

    const int tid   = threadIdx.x;
    const int lane  = tid & 31;
    const int w     = tid >> 5;         // wave id, 0..7
    const int nsub  = lane & 15;        // N (and A-row m) within tile
    const int khalf = lane >> 4;        // lane-half selects K sub-block

    // ---------------- one-time setup ----------------
    // Wcat B-fragments: B[k][j] = Wcat[j][k];  k<7 -> W_ih, 7..31 -> 0, 32.. -> W_hh
    for (int idx = tid; idx < WFRAG_N; idx += 256) {
        int e  = idx & 15;
        int ln = (idx >> 4) & 31;
        int nt = (idx >> 9) & 31;
        int kt = idx >> 14;
        int k  = kt * 32 + ((ln >> 4) << 4) + e;
        int j  = nt * 16 + (ln & 15);
        float v = 0.0f;
        if (k < OUT)       v = W_ih[j * OUT + k];
        else if (k >= 32)  v = W_hh[j * HID + (k - 32)];
        Wfrag[idx] = (__bf16)v;
    }
    // W_out B-fragments: B[k][n] = (n<7) ? W_out[n][k] : 0
    for (int idx = tid; idx < WOFRAG_N; idx += 256) {
        int e  = idx & 15;
        int ln = (idx >> 4) & 31;
        int kt = idx >> 9;
        int k  = kt * 32 + ((ln >> 4) << 4) + e;
        int n  = ln & 15;
        WoFrag[idx] = (__bf16)((n < OUT) ? W_out[n * HID + k] : 0.0f);
    }
    // z init: x0 one-hot at col OUT-1, h part from h0, padding zero
    for (int idx = tid; idx < Z_N; idx += 256) {
        int m = idx / KDIM, k = idx % KDIM;
        float v = 0.0f;
        if (m < BATCH) {
            if (k == OUT - 1)  v = 1.0f;
            else if (k >= 32)  v = h0[m * HID + (k - 32)];
        }
        z[idx] = (__bf16)v;
    }
    for (int idx = tid; idx < 16 * HID; idx += 256) {
        int m = idx / HID, k = idx % HID;
        hS[idx] = (m < BATCH) ? h0[m * HID + k] : 0.0f;
        cS[idx] = (m < BATCH) ? c0[m * HID + k] : 0.0f;
    }
    for (int idx = tid; idx < GATES; idx += 256) biasS[idx] = b_ih[idx] + b_hh[idx];
    for (int idx = tid; idx < 16; idx += 256)    boS[idx]   = (idx < OUT) ? b_out[idx] : 0.0f;
    __syncthreads();

    const int hid = w * 16 + nsub;  // this wave's hidden-column block

    // ---------------- sequential time loop ----------------
    for (int t = 0; t < MAXLEN; ++t) {
        // stage 1: gates = z @ Wcat.T + bias ; wave w owns N-tiles {w, w+8, w+16, w+24}
        v8f acc[4];
        #pragma unroll
        for (int g = 0; g < 4; ++g) {
            float b = biasS[g * HID + hid];
            acc[g] = (v8f){b, b, b, b, b, b, b, b};
        }
        const unsigned int* zw = (const unsigned int*)(z + nsub * KDIM);  // A row m=nsub
        #pragma unroll
        for (int kt = 0; kt < 5; ++kt) {
            AFrag a;
            #pragma unroll
            for (int v = 0; v < 8; ++v) {
                int k = (khalf ? 16 : 0) + 2 * v;
                a.w[v] = zw[kt * 16 + (k >> 1)];
            }
            #pragma unroll
            for (int g = 0; g < 4; ++g) {
                int nt = g * 8 + w;
                BFrag bf;
                const uint4* bp = (const uint4*)(Wfrag + (((kt * 32) + nt) * 32 + lane) * 16);
                bf.q[0] = bp[0]; bf.q[1] = bp[1];
                acc[g] = __builtin_amdgcn_wmma_f32_16x16x32_bf16(
                             false, a.v, false, bf.v, (short)0, acc[g], false, false);
            }
        }
        __syncthreads();  // all reads of z done before h rewrite

        // stage 2: LSTM elementwise; each lane owns 8 (m,hid) cells
        #pragma unroll
        for (int r = 0; r < 8; ++r) {
            int m = r + (khalf ? 8 : 0);
            float ig = sigf(acc[0][r]);
            float fg = sigf(acc[1][r]);
            float gg = tanhfast(acc[2][r]);
            float og = sigf(acc[3][r]);
            float c2 = fg * cS[m * HID + hid] + ig * gg;
            float h2 = og * tanhfast(c2);
            cS[m * HID + hid] = c2;
            hS[m * HID + hid] = h2;
            z[m * KDIM + 32 + hid] = (__bf16)h2;
        }
        __syncthreads();

        // stage 3 (wave 0): logits = h2 @ W_out.T + b_out, softmax, argmax -> next x
        if (w == 0) {
            v8f lac;
            { float b = boS[nsub]; lac = (v8f){b, b, b, b, b, b, b, b}; }
            #pragma unroll
            for (int kt = 0; kt < 4; ++kt) {
                AFrag a;
                #pragma unroll
                for (int v = 0; v < 8; ++v) {
                    int k = (khalf ? 16 : 0) + 2 * v;
                    a.w[v] = zw[16 + kt * 16 + (k >> 1)];   // h starts at col 32 (word 16)
                }
                BFrag bf;
                const uint4* bp = (const uint4*)(WoFrag + (kt * 32 + lane) * 16);
                bf.q[0] = bp[0]; bf.q[1] = bp[1];
                lac = __builtin_amdgcn_wmma_f32_16x16x32_bf16(
                          false, a.v, false, bf.v, (short)0, lac, false, false);
            }
            #pragma unroll
            for (int r = 0; r < 8; ++r) {
                int m = r + (khalf ? 8 : 0);
                logS[m * 16 + nsub] = lac[r];
            }
            asm volatile("s_wait_dscnt 0" ::: "memory");  // wave-internal LDS RAW
            if (lane < BATCH) {
                int b = lane;
                float v[OUT]; float mx = -1e30f; int am = 0;
                #pragma unroll
                for (int o = 0; o < OUT; ++o) {
                    v[o] = logS[b * 16 + o];
                    if (v[o] > mx) { mx = v[o]; am = o; }
                }
                float s = 0.0f;
                #pragma unroll
                for (int o = 0; o < OUT; ++o) s += __expf(v[o] - mx);
                float lse = mx + __logf(s);
                float* op = out + (b * MAXLEN + t) * OUT;
                #pragma unroll
                for (int o = 0; o < OUT; ++o) op[o] = v[o] - lse;
                int tf = tf_mask[t];
                #pragma unroll
                for (int o = 0; o < OUT; ++o) {
                    float nx = tf ? tgt1h[(t + 1) * OUT + o] : ((o == am) ? 1.0f : 0.0f);
                    z[b * KDIM + o] = (__bf16)fmaxf(nx, 0.0f);  // relu(x) for next step
                }
            }
        }
        __syncthreads();
    }

    // final states: out = [log_probs (10*2048*7) | hT (10*128) | cT (10*128)]
    for (int idx = tid; idx < BATCH * HID; idx += 256) {
        out[BATCH * MAXLEN * OUT + idx]               = hS[idx];
        out[BATCH * MAXLEN * OUT + BATCH * HID + idx] = cS[idx];
    }
}

extern "C" void kernel_launch(void* const* d_in, const int* in_sizes, int n_in,
                              void* d_out, int out_size, void* d_ws, size_t ws_size,
                              hipStream_t stream) {
    (void)in_sizes; (void)n_in; (void)d_ws; (void)ws_size; (void)out_size;
    const float* h0     = (const float*)d_in[0];
    const float* c0     = (const float*)d_in[1];
    const float* tgt1h  = (const float*)d_in[2];
    const int*   tfm    = (const int*)  d_in[3];
    const float* W_ih   = (const float*)d_in[4];
    const float* W_hh   = (const float*)d_in[5];
    const float* b_ih   = (const float*)d_in[6];
    const float* b_hh   = (const float*)d_in[7];
    const float* W_out  = (const float*)d_in[8];
    const float* b_out  = (const float*)d_in[9];
    // one persistent workgroup (serial recurrence); 8 waves of 32; ~188 KB LDS
    decoder_lstm_wmma<<<1, 256, SMEM_BYTES, stream>>>(
        h0, c0, tgt1h, tfm, W_ih, W_hh, b_ih, b_hh, W_out, b_out, (float*)d_out);
}
